// Model_12945031430953
// MI455X (gfx1250) — matrix operand order repaired and checked by reading, not compiled, not run on hardware
//
#include <hip/hip_runtime.h>

#define T_LEN 2048

typedef __attribute__((ext_vector_type(16))) __bf16 v16bf;
typedef __attribute__((ext_vector_type(8)))  float  v8f;

union FragBF { uint4 q[2]; v16bf v; };

__device__ __forceinline__ unsigned short f2bf(float f) {
  union { float ff; unsigned int u; } c; c.ff = f;
  unsigned int u = c.u;
  unsigned int r = u + 0x7FFFu + ((u >> 16) & 1u);
  if ((u & 0x7F800000u) == 0x7F800000u) r = u;   // Inf/NaN passthrough
  return (unsigned short)(r >> 16);
}

// ---------------------------------------------------------------------------
// Implicit-GEMM conv1d (stride 1) via WMMA bf16 -> f32, fused bias + GN stats.
//   x:    bf16 [B][I][T]   (packed, batch stride I*T)
//   w2:   bf16 [KW][O][I]  (repacked: A-tile loads contiguous over i)
//   out:  f32  [B][O][T]
//   part: per-block (sum, sumsq) partials for GroupNorm, or nullptr
// Block: 256 threads (8 waves). Tile: 64(O) x 128(T). Wave: 16x64 (4 acc).
// All KW weight sub-tiles staged per 32-channel chunk -> KW*4 WMMAs/barrier.
// ---------------------------------------------------------------------------
#define LDA_ 40
#define LDB_ 40

template <int KW>
__global__ __launch_bounds__(256)
void conv_wmma(const unsigned short* __restrict__ xbf,
               const unsigned short* __restrict__ w2,
               const float* __restrict__ bias,
               float* __restrict__ out,
               float* __restrict__ part,
               int I, int O, int pad)
{
  constexpr int WIN = 128 + KW - 1;
  __shared__ unsigned short lA[KW * 64 * LDA_];
  __shared__ unsigned short lB[WIN * LDB_];
  __shared__ float red[512];

  const int b   = blockIdx.z;
  const int o0  = blockIdx.y * 64;
  const int t0  = blockIdx.x * 128;
  const int tid = threadIdx.x;
  const int lane = tid & 31;
  const int wv   = tid >> 5;
  const int mt   = (wv & 3) * 16;   // wave M offset
  const int nt   = (wv >> 2) * 64;  // wave N offset
  const int half = lane >> 4;
  const int l16  = lane & 15;

  v8f acc[4] = {v8f{}, v8f{}, v8f{}, v8f{}};

  const unsigned short* xb = xbf + (size_t)b * I * T_LEN;

  for (int i0 = 0; i0 < I; i0 += 32) {
    // ---- stage X window transposed: lB[tcol][i'] ----
    {
      int ip = tid >> 3;            // 0..31
      int ii = i0 + ip;
      bool iok = ii < I;
      const unsigned short* row = xb + (size_t)ii * T_LEN;
      if (ii + 32 < I)              // prefetch next channel chunk
        __builtin_prefetch(row + 32 * T_LEN + t0, 0, 0);
      for (int tc = tid & 7; tc < WIN; tc += 8) {
        int t = t0 + tc - pad;
        unsigned short v = 0;
        if (iok && t >= 0 && t < T_LEN) v = row[t];
        lB[tc * LDB_ + ip] = v;
      }
    }
    // ---- stage all KW weight tiles: lA[dk][m][i'] ----
#pragma unroll
    for (int dk = 0; dk < KW; ++dk) {
      int m  = tid >> 2;            // 0..63
      int ib = (tid & 3) * 8;
      int o  = o0 + m;
      const unsigned short* wrow = w2 + ((size_t)dk * O + o) * I + i0;
#pragma unroll
      for (int j = 0; j < 8; ++j) {
        int ii = i0 + ib + j;
        unsigned short v = 0;
        if (o < O && ii < I) v = wrow[ib + j];
        lA[(dk * 64 + m) * LDA_ + ib + j] = v;
      }
    }
    __syncthreads();

#pragma unroll
    for (int dk = 0; dk < KW; ++dk) {
      FragBF fa;
      const unsigned short* abase = &lA[(dk * 64 + mt + l16) * LDA_];
      fa.q[0] = *(const uint4*)(abase + 8 * half);        // K = 8*half .. +7
      fa.q[1] = *(const uint4*)(abase + 16 + 8 * half);   // K = 16+8*half .. +7
#pragma unroll
      for (int g = 0; g < 4; ++g) {
        FragBF fb;
        const unsigned short* bbase = &lB[(nt + g * 16 + l16 + dk) * LDB_ + 8 * half];
        fb.q[0] = *(const uint4*)(bbase);
        fb.q[1] = *(const uint4*)(bbase + 16);
        acc[g] = __builtin_amdgcn_wmma_f32_16x16x32_bf16(false, fa.v, false, fb.v,
                                                         (short)0, acc[g], false, false);
      }
    }
    __syncthreads();
  }

  // ---- store (+bias) and accumulate local GN stats ----
  float s = 0.f, sq = 0.f;
  const int oc0 = o0 + mt + 8 * half;
  const int tc0 = t0 + nt + l16;
#pragma unroll
  for (int r = 0; r < 8; ++r) {
    int oc = oc0 + r;
    if (oc < O) {
      float bz = bias[oc];
      size_t base = ((size_t)b * O + oc) * T_LEN + tc0;
#pragma unroll
      for (int g = 0; g < 4; ++g) {
        float v = acc[g][r] + bz;
        out[base + g * 16] = v;
        s += v; sq += v * v;
      }
    }
  }
  if (part) {
    red[tid] = s; red[256 + tid] = sq;
    __syncthreads();
    for (int st = 128; st > 0; st >>= 1) {
      if (tid < st) {
        red[tid] += red[tid + st];
        red[256 + tid] += red[256 + tid + st];
      }
      __syncthreads();
    }
    if (tid == 0) {
      int nblk = gridDim.x * gridDim.y;
      int blk  = blockIdx.y * gridDim.x + blockIdx.x;
      part[((size_t)b * nblk + blk) * 2 + 0] = red[0];
      part[((size_t)b * nblk + blk) * 2 + 1] = red[256];
    }
  }
}

// fixed-order partial reduction -> stat[b] = (mean, rstd). Deterministic.
__global__ __launch_bounds__(32)
void gn_finalize(const float* __restrict__ part, float* __restrict__ stat,
                 int nblk, float invN)
{
  if (threadIdx.x != 0) return;
  int b = blockIdx.x;
  float s = 0.f, sq = 0.f;
  for (int i = 0; i < nblk; ++i) {
    s  += part[((size_t)b * nblk + i) * 2 + 0];
    sq += part[((size_t)b * nblk + i) * 2 + 1];
  }
  float m = s * invN;
  float v = sq * invN - m * m;
  stat[b * 2 + 0] = m;
  stat[b * 2 + 1] = rsqrtf(v + 1e-5f);
}

// normalize + LeakyReLU(0.02) -> bf16 packed [B][C][T]
__global__ __launch_bounds__(256)
void gn_apply_lrelu(const float* __restrict__ x, const float* __restrict__ stat,
                    const float* __restrict__ g, const float* __restrict__ bt,
                    unsigned short* __restrict__ dst, int C, size_t bstride)
{
  int b = blockIdx.y;
  float m = stat[b * 2], rstd = stat[b * 2 + 1];
  size_t idx = (size_t)blockIdx.x * 256 + threadIdx.x;
  int c = (int)(idx >> 11);
  float v = x[(size_t)b * C * T_LEN + idx];
  v = (v - m) * rstd * g[c] + bt[c];
  v = (v < 0.f) ? 0.02f * v : v;
  dst[(size_t)b * bstride + idx] = f2bf(v);
}

// normalize + GLU (1024 -> 512 channels) -> bf16 channels [0..512) of dst
__global__ __launch_bounds__(256)
void gn_apply_glu(const float* __restrict__ x, const float* __restrict__ stat,
                  const float* __restrict__ g, const float* __restrict__ bt,
                  unsigned short* __restrict__ dst, size_t bstride)
{
  int b = blockIdx.y;
  float m = stat[b * 2], rstd = stat[b * 2 + 1];
  size_t idx = (size_t)blockIdx.x * 256 + threadIdx.x;   // c*T + t, c in [0,512)
  int c = (int)(idx >> 11);
  int t = (int)(idx & (T_LEN - 1));
  size_t base = (size_t)b * 1024 * T_LEN;
  float a  = (x[base + idx] - m) * rstd * g[c] + bt[c];
  float gt = (x[base + (size_t)(c + 512) * T_LEN + t] - m) * rstd * g[c + 512] + bt[c + 512];
  float o = a * (1.f / (1.f + __expf(-gt)));
  dst[(size_t)b * bstride + idx] = f2bf(o);
}

// row-wise L2 normalize (128-dim rows), f32 -> f32
__global__ __launch_bounds__(128)
void norm_rows(const float* __restrict__ src, float* __restrict__ dst)
{
  int row = blockIdx.x;
  int tid = threadIdx.x;
  float v = src[(size_t)row * 128 + tid];
  __shared__ float sh[128];
  sh[tid] = v * v;
  __syncthreads();
  for (int st = 64; st > 0; st >>= 1) {
    if (tid < st) sh[tid] += sh[tid + st];
    __syncthreads();
  }
  float rs = rsqrtf(sh[0] + 1e-12f);
  dst[(size_t)row * 128 + tid] = v * rs;
}

// f32 -> bf16 elementwise
__global__ __launch_bounds__(256)
void cvt_bf16(const float* __restrict__ src, unsigned short* __restrict__ dst, size_t n)
{
  size_t i = (size_t)blockIdx.x * 256 + threadIdx.x;
  if (i < n) dst[i] = f2bf(src[i]);
}

// repack conv weight [O][I][KW] f32 -> [KW][O][I] bf16
__global__ __launch_bounds__(256)
void repack_conv(const float* __restrict__ w, unsigned short* __restrict__ dst,
                 int O, int I, int KW)
{
  size_t n = (size_t)KW * O * I;
  size_t idx = (size_t)blockIdx.x * 256 + threadIdx.x;
  if (idx >= n) return;
  int dk = (int)(idx / ((size_t)O * I));
  size_t rem = idx - (size_t)dk * O * I;
  int o = (int)(rem / I);
  int i = (int)(rem - (size_t)o * I);
  dst[idx] = f2bf(w[((size_t)o * I + i) * KW + dk]);
}

// repack convT weight [I][O][KW] f32 -> equivalent-conv [KW][O][I] bf16 (flip k)
__global__ __launch_bounds__(256)
void repack_convT(const float* __restrict__ w, unsigned short* __restrict__ dst,
                  int O, int I, int KW)
{
  size_t n = (size_t)KW * O * I;
  size_t idx = (size_t)blockIdx.x * 256 + threadIdx.x;
  if (idx >= n) return;
  int dk = (int)(idx / ((size_t)O * I));
  size_t rem = idx - (size_t)dk * O * I;
  int o = (int)(rem / I);
  int i = (int)(rem - (size_t)o * I);
  dst[idx] = f2bf(w[((size_t)i * O + o) * KW + (KW - 1 - dk)]);
}

// ---------------------------------------------------------------------------
// VQ: per (b,t) l2norm(z) . cbn[k], argmax over 512 codes; write code bf16
// into channels [0..128) of dst (batch stride bstride). One wave per t.
// ---------------------------------------------------------------------------
__global__ __launch_bounds__(256)
void vq_kernel(const float* __restrict__ z, const float* __restrict__ cbn,
               unsigned short* __restrict__ dst, size_t bstride)
{
  __shared__ float zn[8][128];
  int b = blockIdx.y;
  int wv = threadIdx.x >> 5, lane = threadIdx.x & 31;
  int t = blockIdx.x * 8 + wv;
  const float* zb = z + (size_t)b * 128 * T_LEN + t;

  float v0 = zb[(size_t)(lane * 4 + 0) * T_LEN];
  float v1 = zb[(size_t)(lane * 4 + 1) * T_LEN];
  float v2 = zb[(size_t)(lane * 4 + 2) * T_LEN];
  float v3 = zb[(size_t)(lane * 4 + 3) * T_LEN];
  float ss = v0 * v0 + v1 * v1 + v2 * v2 + v3 * v3;
  for (int m = 16; m >= 1; m >>= 1) ss += __shfl_xor(ss, m, 32);
  float rs = rsqrtf(ss + 1e-12f);
  zn[wv][lane * 4 + 0] = v0 * rs;
  zn[wv][lane * 4 + 1] = v1 * rs;
  zn[wv][lane * 4 + 2] = v2 * rs;
  zn[wv][lane * 4 + 3] = v3 * rs;
  __syncthreads();

  float best = -1e30f; int bi = 0x7fffffff;
  for (int kk = 0; kk < 16; ++kk) {
    int k = lane * 16 + kk;
    const float* cr = cbn + (size_t)k * 128;
    float d = 0.f;
#pragma unroll 4
    for (int j = 0; j < 128; ++j) d += zn[wv][j] * cr[j];
    if (d > best) { best = d; bi = k; }
  }
  for (int m = 16; m >= 1; m >>= 1) {
    float ov = __shfl_xor(best, m, 32);
    int   oi = __shfl_xor(bi, m, 32);
    if (ov > best || (ov == best && oi < bi)) { best = ov; bi = oi; }
  }
  const float* cw = cbn + (size_t)bi * 128;
  unsigned short* dp = dst + (size_t)b * bstride + t;
#pragma unroll
  for (int j = 0; j < 4; ++j) {
    int d = lane * 4 + j;
    dp[(size_t)d * T_LEN] = f2bf(cw[d]);
  }
}

// splat normalized speaker embedding rows into channels [coff..coff+128) of dst
__global__ __launch_bounds__(256)
void fill_yv(const int* __restrict__ y, const float* __restrict__ spkn,
             unsigned short* __restrict__ dst, int coff, size_t bstride)
{
  int b = blockIdx.y;
  int t = blockIdx.x * 256 + threadIdx.x;
  int yi = y[(size_t)b * T_LEN + t];
  const float* e = spkn + (size_t)yi * 128;
  unsigned short* dp = dst + (size_t)b * bstride + (size_t)coff * T_LEN + t;
  for (int d = 0; d < 128; ++d) dp[(size_t)d * T_LEN] = f2bf(e[d]);
}

// ---------------------------------------------------------------------------
extern "C" void kernel_launch(void* const* d_in, const int* in_sizes, int n_in,
                              void* d_out, int out_size, void* d_ws, size_t ws_size,
                              hipStream_t stream)
{
  (void)in_sizes; (void)n_in; (void)out_size; (void)ws_size;
  const float* x    = (const float*)d_in[0];
  const int*   y    = (const int*)d_in[1];
  const float* spk  = (const float*)d_in[2];
  const float* cb   = (const float*)d_in[3];
  const float* ew0  = (const float*)d_in[4];  const float* eb0 = (const float*)d_in[5];
  const float* eg0  = (const float*)d_in[6];  const float* et0 = (const float*)d_in[7];
  const float* ew1  = (const float*)d_in[8];  const float* eb1 = (const float*)d_in[9];
  const float* eg1  = (const float*)d_in[10]; const float* et1 = (const float*)d_in[11];
  const float* ew2  = (const float*)d_in[12]; const float* eb2 = (const float*)d_in[13];
  const float* eg2  = (const float*)d_in[14]; const float* et2 = (const float*)d_in[15];
  const float* mw   = (const float*)d_in[16]; const float* mb  = (const float*)d_in[17];
  const float* dw0  = (const float*)d_in[18]; const float* db0 = (const float*)d_in[19];
  const float* dg0  = (const float*)d_in[20]; const float* dt0 = (const float*)d_in[21];
  const float* dw1  = (const float*)d_in[22]; const float* db1 = (const float*)d_in[23];
  const float* dg1  = (const float*)d_in[24]; const float* dt1 = (const float*)d_in[25];
  const float* dw2  = (const float*)d_in[26]; const float* db2 = (const float*)d_in[27];
  float* out = (float*)d_out;

  const int B = 16;
  char* ws = (char*)d_ws;
  size_t o_xbf0 = 0;
  size_t o_xbf1 = o_xbf0 + (size_t)B * 1024 * T_LEN * 2;
  size_t o_f32  = o_xbf1 + (size_t)B * 1024 * T_LEN * 2;
  size_t o_w2   = o_f32  + (size_t)B * 1024 * T_LEN * 4;
  size_t o_spkn = o_w2   + (size_t)8 * 1024 * 1024;
  size_t o_cbn  = o_spkn + 64 * 128 * 4;
  size_t o_part = o_cbn  + 512 * 128 * 4;
  size_t o_stat = o_part + (size_t)B * 256 * 2 * 4;
  size_t o_z    = o_stat + 1024;

  unsigned short* XBF0 = (unsigned short*)(ws + o_xbf0);
  unsigned short* XBF1 = (unsigned short*)(ws + o_xbf1);
  float*          F32  = (float*)(ws + o_f32);
  unsigned short* W2   = (unsigned short*)(ws + o_w2);
  float*          SPKN = (float*)(ws + o_spkn);
  float*          CBN  = (float*)(ws + o_cbn);
  float*          PART = (float*)(ws + o_part);
  float*          STAT = (float*)(ws + o_stat);
  float*          Z    = (float*)(ws + o_z);

  dim3 blk(256);
  auto gr1 = [](size_t n) { return dim3((unsigned)((n + 255) / 256)); };
  const float invN1024 = 1.f / (1024.f * T_LEN);
  const float invN512  = 1.f / (512.f * T_LEN);

  // --- prep: normalized speaker emb + codebook, x -> bf16 ---
  norm_rows<<<64,  128, 0, stream>>>(spk, SPKN);
  norm_rows<<<512, 128, 0, stream>>>(cb, CBN);
  cvt_bf16<<<gr1((size_t)B * 39 * T_LEN), blk, 0, stream>>>(x, XBF0, (size_t)B * 39 * T_LEN);

  // --- encoder layer 0: 39 -> 1024 ---
  repack_conv<<<gr1((size_t)5 * 1024 * 39), blk, 0, stream>>>(ew0, W2, 1024, 39, 5);
  conv_wmma<5><<<dim3(16, 16, B), blk, 0, stream>>>(XBF0, W2, eb0, F32, PART, 39, 1024, 2);
  gn_finalize<<<B, 32, 0, stream>>>(PART, STAT, 256, invN1024);
  gn_apply_lrelu<<<dim3(8192, B), blk, 0, stream>>>(F32, STAT, eg0, et0, XBF1, 1024, (size_t)1024 * T_LEN);

  // --- encoder layer 1: 1024 -> 512 ---
  repack_conv<<<gr1((size_t)5 * 512 * 1024), blk, 0, stream>>>(ew1, W2, 512, 1024, 5);
  conv_wmma<5><<<dim3(16, 8, B), blk, 0, stream>>>(XBF1, W2, eb1, F32, PART, 1024, 512, 2);
  gn_finalize<<<B, 32, 0, stream>>>(PART, STAT, 128, invN512);
  gn_apply_lrelu<<<dim3(4096, B), blk, 0, stream>>>(F32, STAT, eg1, et1, XBF0, 512, (size_t)512 * T_LEN);

  // --- encoder layer 2: 512 -> 512 ---
  repack_conv<<<gr1((size_t)5 * 512 * 512), blk, 0, stream>>>(ew2, W2, 512, 512, 5);
  conv_wmma<5><<<dim3(16, 8, B), blk, 0, stream>>>(XBF0, W2, eb2, F32, PART, 512, 512, 2);
  gn_finalize<<<B, 32, 0, stream>>>(PART, STAT, 128, invN512);
  gn_apply_lrelu<<<dim3(4096, B), blk, 0, stream>>>(F32, STAT, eg2, et2, XBF1, 512, (size_t)512 * T_LEN);

  // --- 1x1 projection: 512 -> 128 (z) ---
  repack_conv<<<gr1((size_t)128 * 512), blk, 0, stream>>>(mw, W2, 128, 512, 1);
  conv_wmma<1><<<dim3(16, 2, B), blk, 0, stream>>>(XBF1, W2, mb, Z, nullptr, 512, 128, 0);

  // --- VQ + speaker concat into XBF0 [B][256][T] ---
  vq_kernel<<<dim3(T_LEN / 8, B), blk, 0, stream>>>(Z, CBN, XBF0, (size_t)256 * T_LEN);
  fill_yv<<<dim3(T_LEN / 256, B), blk, 0, stream>>>(y, SPKN, XBF0, 128, (size_t)256 * T_LEN);

  // --- decoder layer 0: 256 -> 1024, GN, GLU -> 512 ; concat yv -> 640 ---
  repack_convT<<<gr1((size_t)5 * 1024 * 256), blk, 0, stream>>>(dw0, W2, 1024, 256, 5);
  conv_wmma<5><<<dim3(16, 16, B), blk, 0, stream>>>(XBF0, W2, db0, F32, PART, 256, 1024, 2);
  gn_finalize<<<B, 32, 0, stream>>>(PART, STAT, 256, invN1024);
  gn_apply_glu<<<dim3(4096, B), blk, 0, stream>>>(F32, STAT, dg0, dt0, XBF1, (size_t)640 * T_LEN);
  fill_yv<<<dim3(T_LEN / 256, B), blk, 0, stream>>>(y, SPKN, XBF1, 512, (size_t)640 * T_LEN);

  // --- decoder layer 1: 640 -> 1024, GN, GLU -> 512 ; concat yv -> 640 ---
  repack_convT<<<gr1((size_t)5 * 1024 * 640), blk, 0, stream>>>(dw1, W2, 1024, 640, 5);
  conv_wmma<5><<<dim3(16, 16, B), blk, 0, stream>>>(XBF1, W2, db1, F32, PART, 640, 1024, 2);
  gn_finalize<<<B, 32, 0, stream>>>(PART, STAT, 256, invN1024);
  gn_apply_glu<<<dim3(4096, B), blk, 0, stream>>>(F32, STAT, dg1, dt1, XBF0, (size_t)640 * T_LEN);
  fill_yv<<<dim3(T_LEN / 256, B), blk, 0, stream>>>(y, SPKN, XBF0, 512, (size_t)640 * T_LEN);

  // --- decoder layer 2: 640 -> 39 (final, straight to d_out) ---
  repack_convT<<<gr1((size_t)5 * 39 * 640), blk, 0, stream>>>(dw2, W2, 39, 640, 5);
  conv_wmma<5><<<dim3(16, 1, B), blk, 0, stream>>>(XBF0, W2, db2, out, nullptr, 640, 39, 2);
}
